// AoASCAttEnc_50586124812563
// MI455X (gfx1250) — compile-verified, matrix operand order checked by reading, not run on hardware
//
#include <hip/hip_runtime.h>
#include <hip/hip_bf16.h>
#include <stdint.h>

// ---------------------------------------------------------------------------
// Fused AoA spatial-attention + channel-gate kernel for MI455X (gfx1250).
// Flash-attention style, bf16 WMMA, f32 accumulate. Register-double-buffered
// K/V staging (vectorized b128 loads + packed bf16 LDS stores), async-to-LDS
// mask staging (global_load_async_to_lds_b32 + s_wait_asynccnt).
// ---------------------------------------------------------------------------

typedef __attribute__((ext_vector_type(16))) __bf16 v16bf;
typedef __attribute__((ext_vector_type(8)))  float  v8f;

#define BDIM 8
#define HDIM 8
#define SDIM 1024
#define DDIM 64
#define QT   128                // query rows per block (8 waves x 16 rows)
#define KT   64                 // key rows per tile
#define NT   (SDIM / KT)
#define NEG_INF_F (-1000000000.0f)

// Forward map: 16-bit A/B fragment element j (0..15) + lane-half -> k (0..31)
// (CDNA5 ISA 7.12.2, 16-bit 16x32 layout)
__device__ __forceinline__ int frag_k(int j, int hf) {
    int p = j >> 1;
    return (p < 4) ? (p * 2 + (j & 1) + hf * 8)
                   : (16 + (p - 4) * 2 + (j & 1) + hf * 8);
}

// Inverse map: k (0..31) -> (lane-half, element j). j is CONTIGUOUS in k
// within each aligned 4-run, which is what makes staging vectorizable.
__device__ __forceinline__ void frag_inv(int kk, int& hf, int& j) {
    if (kk < 16) { hf = kk >> 3; j = kk & 7; }
    else         { hf = (kk - 16) >> 3; j = 8 + ((kk - 16) & 7); }
}

// ---- double-buffered tile staging -----------------------------------------
// Thread t owns 4 runs (rid = t + i*256) of 4 consecutive elements:
//   K: (krow = rid>>4, d    = (rid&15)*4 .. +3)   -> float4 global load
//   V: (e    = rid>>4, krow = (rid&15)*4 .. +3)   -> 4 strided b32 loads
// Both land on 4 consecutive j slots of one fragment lane -> 8B LDS store.
__device__ __forceinline__ void stage_global(const float* __restrict__ key,
                                             const float* __restrict__ v2,
                                             size_t hbase, int kbase, int t,
                                             float* kreg, float* vreg)
{
    #pragma unroll
    for (int i = 0; i < 4; ++i) {
        const int rid = t + i * 256;
        const int r16 = rid >> 4;
        const int c4  = (rid & 15) << 2;
        const float4 kq =
            *(const float4*)&key[hbase + (size_t)(kbase + r16) * DDIM + c4];
        kreg[i*4+0] = kq.x; kreg[i*4+1] = kq.y;
        kreg[i*4+2] = kq.z; kreg[i*4+3] = kq.w;
        #pragma unroll
        for (int q = 0; q < 4; ++q)
            vreg[i*4+q] = v2[hbase + (size_t)(kbase + c4 + q) * DDIM + r16];
    }
}

__device__ __forceinline__ void stage_lds(int buf, int t,
                                          const float* kreg, const float* vreg,
                                          __bf16* ldsK, __bf16* ldsV)
{
    #pragma unroll
    for (int i = 0; i < 4; ++i) {
        const int rid = t + i * 256;
        const int r16 = rid >> 4;
        const int c4  = (rid & 15) << 2;
        int hf, jb;
        frag_inv(c4 & 31, hf, jb);
        const int f  = ((c4 >> 5) << 2) | (r16 >> 4);   // kc*4 + nc
        const int fl = hf * 16 + (r16 & 15);
        {   // K fragment: B[d][n] = K[krow][d]
            union { __bf16 h[4]; uint2 u; } pk;
            pk.h[0] = (__bf16)kreg[i*4+0]; pk.h[1] = (__bf16)kreg[i*4+1];
            pk.h[2] = (__bf16)kreg[i*4+2]; pk.h[3] = (__bf16)kreg[i*4+3];
            *(uint2*)&ldsK[buf*4096 + f*512 + fl*16 + jb] = pk.u;
        }
        {   // V fragment: B[k][e] = V2[krow][e]
            union { __bf16 h[4]; uint2 u; } pv;
            pv.h[0] = (__bf16)vreg[i*4+0]; pv.h[1] = (__bf16)vreg[i*4+1];
            pv.h[2] = (__bf16)vreg[i*4+2]; pv.h[3] = (__bf16)vreg[i*4+3];
            *(uint2*)&ldsV[buf*4096 + f*512 + fl*16 + jb] = pv.u;
        }
    }
}

// CDNA5 async global->LDS copy (raw bytes, tracked by ASYNCcnt).
// LDS offset = low 32 bits of the generic shared-pointer (ISA 10.2).
__device__ __forceinline__ void async_mask_load(const int* gp, int* lp)
{
    const unsigned           lds_off = (unsigned)(uintptr_t)lp;
    const unsigned long long ga      = (unsigned long long)(uintptr_t)gp;
    asm volatile("global_load_async_to_lds_b32 %0, %1, off"
                 :: "v"(lds_off), "v"(ga) : "memory");
}
__device__ __forceinline__ void wait_async0()
{
    asm volatile("s_wait_asynccnt 0x0" ::: "memory");
}

__global__ __launch_bounds__(256)
void aoa_attn_kernel(const float* __restrict__ q2,
                     const float* __restrict__ key,
                     const int*   __restrict__ att_mask,
                     const float* __restrict__ v1,
                     const float* __restrict__ v2,
                     const float* __restrict__ W_ch,
                     const float* __restrict__ b_ch,
                     float* __restrict__ out)
{
    __shared__ __bf16 ldsK[2 * 4096];   // double-buffered K B-fragments
    __shared__ __bf16 ldsV[2 * 4096];   // double-buffered V2 B-fragments
    __shared__ __bf16 ldsW[4096];       // W_ch B-fragments
    __shared__ __bf16 ldsP[8 * 1024];   // per-wave P transpose scratch
    __shared__ int    ldsM[2 * KT];     // double-buffered mask tile (async)

    const int t    = threadIdx.x;
    const int lane = t & 31;
    const int wv   = t >> 5;
    const int col  = lane & 15;
    const int hf   = lane >> 4;

    const int bh    = blockIdx.y;
    const int b     = bh >> 3;
    const int qtile = blockIdx.x;
    const int qrowA = qtile * QT + wv * 16 + col;
    const size_t hbase = (size_t)bh * SDIM * DDIM;

    // ---- Stage W_ch into B-fragment layout once (8 frags == 8 waves) ----
    {
        const int f = wv, fl = lane;
        const int nc = f & 3, kc = f >> 2;
        #pragma unroll
        for (int j = 0; j < 16; ++j) {
            int e = nc * 16 + (fl & 15);
            int d = kc * 32 + frag_k(j, fl >> 4);
            ldsW[f * 512 + fl * 16 + j] = (__bf16)W_ch[e * DDIM + d];
        }
    }

    // ---- Load Q and (V1 .* K) A-fragments for this wave's 16 rows ----
    v16bf aQ[2], aAM[2];
    {
        const size_t rb = ((size_t)bh * SDIM + qrowA) * DDIM;
        #pragma unroll
        for (int kc = 0; kc < 2; ++kc)
            #pragma unroll
            for (int j = 0; j < 16; ++j) {
                int d = kc * 32 + frag_k(j, hf);
                aQ[kc][j]  = (__bf16)q2[rb + d];
                aAM[kc][j] = (__bf16)(v1[rb + d] * key[rb + d]);
            }
    }

    // ---- Online-softmax state ----
    v8f acc[4], mrow, lrow;
    #pragma unroll
    for (int r = 0; r < 8; ++r) { mrow[r] = -3.0e38f; lrow[r] = 0.0f; }
    #pragma unroll
    for (int dc = 0; dc < 4; ++dc)
        #pragma unroll
        for (int r = 0; r < 8; ++r) acc[dc][r] = 0.0f;

    // ---- prologue: stage tile 0 ----
    float kreg[16], vreg[16];
    if (t < KT) async_mask_load(&att_mask[b * SDIM + t], &ldsM[t]);
    stage_global(key, v2, hbase, 0, t, kreg, vreg);
    stage_lds(0, t, kreg, vreg, ldsK, ldsV);
    wait_async0();
    __syncthreads();

    // =============== main loop over K tiles ===============
    for (int kt = 0; kt < NT; ++kt) {
        const int buf   = kt & 1;
        const int kbase = kt * KT;

        // issue next tile's global traffic up front (hidden behind WMMA)
        if (kt + 1 < NT) {
            if (t < KT)
                async_mask_load(&att_mask[b * SDIM + kbase + KT + t],
                                &ldsM[(buf ^ 1) * KT + t]);
            stage_global(key, v2, hbase, kbase + KT, t, kreg, vreg);
        }

        // -- scores tile: 16 x 64, via 4 x (2 WMMA) --
        v8f sc[4];
        #pragma unroll
        for (int nc = 0; nc < 4; ++nc) {
            v8f c;
            #pragma unroll
            for (int r = 0; r < 8; ++r) c[r] = 0.0f;
            #pragma unroll
            for (int kc = 0; kc < 2; ++kc) {
                v16bf bK = *(const v16bf*)
                    &ldsK[buf*4096 + (kc * 4 + nc) * 512 + lane * 16];
                c = __builtin_amdgcn_wmma_f32_16x16x32_bf16(
                        false, aQ[kc], false, bK, (short)0, c, false, false);
            }
            const int mv = ldsM[buf * KT + nc * 16 + col];
            #pragma unroll
            for (int r = 0; r < 8; ++r)
                sc[nc][r] = mv ? c[r] * 0.125f : NEG_INF_F;
        }

        // -- row max across the 16-lane half-group --
        v8f mnew = sc[0];
        #pragma unroll
        for (int nc = 1; nc < 4; ++nc)
            #pragma unroll
            for (int r = 0; r < 8; ++r) mnew[r] = fmaxf(mnew[r], sc[nc][r]);
        #pragma unroll
        for (int s = 1; s < 16; s <<= 1)
            #pragma unroll
            for (int r = 0; r < 8; ++r)
                mnew[r] = fmaxf(mnew[r], __shfl_xor(mnew[r], s, 32));

        v8f corr;
        #pragma unroll
        for (int r = 0; r < 8; ++r) {
            float mt = fmaxf(mrow[r], mnew[r]);
            corr[r]  = __expf(mrow[r] - mt);
            mrow[r]  = mt;
        }

        // -- P = exp(sc - m), row sums --
        v8f rsum;
        #pragma unroll
        for (int r = 0; r < 8; ++r) rsum[r] = 0.0f;
        #pragma unroll
        for (int nc = 0; nc < 4; ++nc)
            #pragma unroll
            for (int r = 0; r < 8; ++r) {
                float p = __expf(sc[nc][r] - mrow[r]);
                sc[nc][r] = p;
                rsum[r] += p;
            }
        #pragma unroll
        for (int s = 1; s < 16; s <<= 1)
            #pragma unroll
            for (int r = 0; r < 8; ++r)
                rsum[r] += __shfl_xor(rsum[r], s, 32);
        #pragma unroll
        for (int r = 0; r < 8; ++r) lrow[r] = lrow[r] * corr[r] + rsum[r];
        #pragma unroll
        for (int dc = 0; dc < 4; ++dc)
            #pragma unroll
            for (int r = 0; r < 8; ++r) acc[dc][r] *= corr[r];

        // -- C-layout -> A-layout transpose of P through per-wave LDS --
        const int pbase = wv * 1024;
        #pragma unroll
        for (int nc = 0; nc < 4; ++nc) {
            int ck = nc * 16 + col;
            int kc = ck >> 5, kk = ck & 31;
            int hf2, j2;
            frag_inv(kk, hf2, j2);
            #pragma unroll
            for (int r = 0; r < 8; ++r) {
                int row = hf * 8 + r;
                ldsP[pbase + kc * 512 + (hf2 * 16 + row) * 16 + j2] =
                    (__bf16)sc[nc][r];
            }
        }
        v16bf pA0 = *(const v16bf*)&ldsP[pbase + lane * 16];
        v16bf pA1 = *(const v16bf*)&ldsP[pbase + 512 + lane * 16];

        // -- accumulate O += P x V2 : 4 x (2 WMMA) --
        #pragma unroll
        for (int dc = 0; dc < 4; ++dc) {
            v16bf bV0 = *(const v16bf*)&ldsV[buf*4096 + (0 + dc) * 512 + lane * 16];
            v16bf bV1 = *(const v16bf*)&ldsV[buf*4096 + (4 + dc) * 512 + lane * 16];
            acc[dc] = __builtin_amdgcn_wmma_f32_16x16x32_bf16(
                          false, pA0, false, bV0, (short)0, acc[dc], false, false);
            acc[dc] = __builtin_amdgcn_wmma_f32_16x16x32_bf16(
                          false, pA1, false, bV1, (short)0, acc[dc], false, false);
        }

        // convert + publish next tile, single barrier per iteration
        if (kt + 1 < NT)
            stage_lds(buf ^ 1, t, kreg, vreg, ldsK, ldsV);
        wait_async0();
        __syncthreads();
    }

    // =============== epilogue: normalize + channel gate ===============
    v8f linv;
    #pragma unroll
    for (int r = 0; r < 8; ++r) linv[r] = 1.0f / lrow[r];

    const size_t obase = 512 + ((size_t)bh * SDIM + qtile * QT + wv * 16) * DDIM;
    #pragma unroll
    for (int nc = 0; nc < 4; ++nc) {
        v8f g;
        #pragma unroll
        for (int r = 0; r < 8; ++r) g[r] = 0.0f;
        #pragma unroll
        for (int kc = 0; kc < 2; ++kc) {
            v16bf bW = *(const v16bf*)&ldsW[(kc * 4 + nc) * 512 + lane * 16];
            g = __builtin_amdgcn_wmma_f32_16x16x32_bf16(
                    false, aAM[kc], false, bW, (short)0, g, false, false);
        }
        const int e    = nc * 16 + col;
        const float bc = b_ch[e];
        #pragma unroll
        for (int r = 0; r < 8; ++r) {
            int   row  = hf * 8 + r;
            float gate = 1.0f / (1.0f + __expf(-(g[r] + bc)));
            out[obase + (size_t)row * DDIM + e] = gate * acc[nc][r] * linv[r];
        }
    }
}

extern "C" void kernel_launch(void* const* d_in, const int* in_sizes, int n_in,
                              void* d_out, int out_size, void* d_ws, size_t ws_size,
                              hipStream_t stream) {
    const float* gv_feat = (const float*)d_in[0];  // (8, 64)
    const float* q2      = (const float*)d_in[1];  // (8, 8, 1024, 64)
    const float* key     = (const float*)d_in[2];  // (8, 8, 1024, 64)
    const int*   amask   = (const int*)  d_in[3];  // (8, 1024)
    const float* v1      = (const float*)d_in[4];  // (8, 8, 1024, 64)
    const float* v2      = (const float*)d_in[5];  // (8, 8, 1024, 64)
    const float* W_ch    = (const float*)d_in[6];  // (64, 64)
    const float* b_ch    = (const float*)d_in[7];  // (64,)
    float* out = (float*)d_out;

    hipMemcpyAsync(out, gv_feat, (size_t)BDIM * DDIM * sizeof(float),
                   hipMemcpyDeviceToDevice, stream);

    dim3 grid(SDIM / QT, BDIM * HDIM);
    dim3 block(256);
    aoa_attn_kernel<<<grid, block, 0, stream>>>(q2, key, amask, v1, v2,
                                                W_ch, b_ch, out);
}
